// GPT_55679956025773
// MI455X (gfx1250) — matrix operand order repaired and checked by reading, not compiled, not run on hardware
//
#include <hip/hip_runtime.h>
#include <hip/hip_bf16.h>
#include <math.h>

typedef __bf16 bf16;
typedef __attribute__((ext_vector_type(16))) __bf16 v16bf;
typedef __attribute__((ext_vector_type(8)))  float  v8f;
typedef __attribute__((ext_vector_type(4)))  int    v4i;

typedef __attribute__((address_space(1))) void* gptr_t;  // global void*
typedef __attribute__((address_space(3))) void* lptr_t;  // LDS void*
typedef __attribute__((address_space(1))) v4i*  g4p_t;   // global int4*
typedef __attribute__((address_space(3))) v4i*  l4p_t;   // LDS int4*

// Problem constants
static constexpr int BB   = 4;
static constexpr int TT   = 1024;
static constexpr int DM   = 1024;
static constexpr int HH   = 16;
static constexpr int DH   = 64;
static constexpr int LL   = 6;
static constexpr int FF   = 4096;    // 4*D
static constexpr int OUTD = 1024;
static constexpr int MR   = BB * TT; // 4096 rows

// ---------------------------------------------------------------------------
// CDNA5 async global->LDS staging (ASYNCcnt path), with sync fallback
// ---------------------------------------------------------------------------
#if defined(__has_builtin)
#if __has_builtin(__builtin_amdgcn_global_load_async_to_lds_b128)
#define USE_ASYNC 1
#endif
#endif
#ifndef USE_ASYNC
#define USE_ASYNC 0
#endif

__device__ __forceinline__ void stage_b128(const bf16* g, bf16* l) {
#if USE_ASYNC
  g4p_t gs = (g4p_t)(gptr_t)(void*)const_cast<bf16*>(g);
  l4p_t ls = (l4p_t)(lptr_t)(void*)l;
  __builtin_amdgcn_global_load_async_to_lds_b128(gs, ls, 0, 0);
#else
  *reinterpret_cast<uint4*>(l) = *reinterpret_cast<const uint4*>(g);
#endif
}

__device__ __forceinline__ void wait_stage() {
#if USE_ASYNC
#if __has_builtin(__builtin_amdgcn_s_wait_asynccnt)
  __builtin_amdgcn_s_wait_asynccnt(0);
#else
  asm volatile("s_wait_asynccnt 0" ::: "memory");
#endif
#endif
}

__device__ __forceinline__ v16bf frag2(const bf16* p0, const bf16* p1) {
  union { uint4 u; bf16 e[8]; } a, b;
  a.u = *reinterpret_cast<const uint4*>(p0);
  b.u = *reinterpret_cast<const uint4*>(p1);
  v16bf f;
#pragma unroll
  for (int i = 0; i < 8; ++i) { f[i] = a.e[i]; f[i + 8] = b.e[i]; }
  return f;
}

// ---------------------------------------------------------------------------
// x = seq + pos_emb (broadcast over batch)
// ---------------------------------------------------------------------------
__global__ void addpos_kernel(const float* __restrict__ seq,
                              const float* __restrict__ pos,
                              float* __restrict__ x, long total) {
  long i = (long)blockIdx.x * blockDim.x + threadIdx.x;
  if (i < total) x[i] = seq[i] + pos[i & (long)(TT * DM - 1)];
}

// ---------------------------------------------------------------------------
// Weight convert + transpose: in f32 [K,N] -> out bf16 [N,K]
// ---------------------------------------------------------------------------
__global__ __launch_bounds__(256)
void cvtT_kernel(const float* __restrict__ in, bf16* __restrict__ out,
                 int K, int N) {
  __shared__ bf16 t[32][40];
  const int n0 = blockIdx.x * 32, k0 = blockIdx.y * 32;
  const int tx = threadIdx.x & 31, ty = threadIdx.x >> 5;  // ty: 0..7
#pragma unroll
  for (int j = 0; j < 4; ++j)
    t[tx][ty + j * 8] = (bf16)in[(size_t)(k0 + ty + j * 8) * N + n0 + tx];
  __syncthreads();
#pragma unroll
  for (int j = 0; j < 4; ++j)
    out[(size_t)(n0 + ty + j * 8) * K + k0 + tx] = t[ty + j * 8][tx];
}

// V transpose per head: v bf16 [B*T, D] -> vt bf16 [B,H,DH,T]
__global__ __launch_bounds__(256)
void vtrans_kernel(const bf16* __restrict__ v, bf16* __restrict__ vt) {
  const int bh = blockIdx.z, b = bh >> 4, hh = bh & 15;
  __shared__ bf16 t[32][40];
  const int t0 = blockIdx.x * 32, d0 = blockIdx.y * 32;
  const int tx = threadIdx.x & 31, ty = threadIdx.x >> 5;
#pragma unroll
  for (int j = 0; j < 4; ++j)
    t[tx][ty + j * 8] =
        v[((size_t)b * TT + t0 + ty + j * 8) * DM + hh * DH + d0 + tx];
  __syncthreads();
#pragma unroll
  for (int j = 0; j < 4; ++j)
    vt[((size_t)bh * DH + d0 + ty + j * 8) * TT + t0 + tx] = t[ty + j * 8][tx];
}

// ---------------------------------------------------------------------------
// LayerNorm: one row (D=1024) per 256-thread block, writes bf16
// ---------------------------------------------------------------------------
__global__ __launch_bounds__(256)
void ln_kernel(const float* __restrict__ x, const float* __restrict__ g,
               const float* __restrict__ b, bf16* __restrict__ out) {
  const int row = blockIdx.x;
  const float* xr = x + (size_t)row * DM;
  __shared__ float red[256];
  float s = 0.f;
  for (int i = threadIdx.x; i < DM; i += 256) s += xr[i];
  red[threadIdx.x] = s; __syncthreads();
  for (int o = 128; o > 0; o >>= 1) {
    if (threadIdx.x < o) red[threadIdx.x] += red[threadIdx.x + o];
    __syncthreads();
  }
  float mu = red[0] / DM; __syncthreads();
  float vs = 0.f;
  for (int i = threadIdx.x; i < DM; i += 256) { float d = xr[i] - mu; vs += d * d; }
  red[threadIdx.x] = vs; __syncthreads();
  for (int o = 128; o > 0; o >>= 1) {
    if (threadIdx.x < o) red[threadIdx.x] += red[threadIdx.x + o];
    __syncthreads();
  }
  float rstd = rsqrtf(red[0] / DM + 1e-5f);
  bf16* orow = out + (size_t)row * DM;
  for (int i = threadIdx.x; i < DM; i += 256)
    orow[i] = (bf16)((xr[i] - mu) * rstd * g[i] + b[i]);
}

// ---------------------------------------------------------------------------
// GEMM: C[M,N] = epilogue(A[M,K](bf16) * WT[N,K](bf16)^T + bias)
// block = 256 threads (8 waves), tile 128x64, BK=32, double-buffered LDS with
// async global->LDS prefetch. Wave computes 32x32 (4 WMMA / K-step).
// MODE 0: bf16 out; 1: exact GELU -> bf16; 2: f32 out = resid + val; 3: f32 out
// ---------------------------------------------------------------------------
template <int MODE>
__global__ __launch_bounds__(256)
void gemm_kernel(const bf16* __restrict__ A, const bf16* __restrict__ WT,
                 const float* __restrict__ bias, void* outv, const float* resid,
                 int M, int N, int K) {
  __shared__ bf16 As[2][128][40];  // stride 40: 16B-aligned b128 LDS ops
  __shared__ bf16 Bs[2][64][40];   // [n][k] (WT is already transposed)
  const int tid = threadIdx.x, w = tid >> 5, lane = tid & 31;
  const int l16 = lane & 15, hl = lane >> 4;
  const int wm = w & 3, wn = w >> 2;
  const int row0 = blockIdx.y * 128, n0 = blockIdx.x * 64;
  const int ar = tid >> 1, acol = (tid & 1) * 16;  // A: 128 rows x 32
  const int bn = tid >> 2, bcol = (tid & 3) * 8;   // B: 64 rows x 32

  auto stage = [&](int buf, int k0) {
    const bf16* ga = &A[(size_t)(row0 + ar) * K + k0 + acol];
    stage_b128(ga,     &As[buf][ar][acol]);
    stage_b128(ga + 8, &As[buf][ar][acol + 8]);
    stage_b128(&WT[(size_t)(n0 + bn) * K + k0 + bcol], &Bs[buf][bn][bcol]);
  };

  v8f z = {};
  v8f acc[2][2];
  acc[0][0] = z; acc[0][1] = z; acc[1][0] = z; acc[1][1] = z;

  stage(0, 0);
  wait_stage();
  __syncthreads();

  int buf = 0;
  for (int k0 = 0; k0 < K; k0 += 32) {
    if (k0 + 32 < K) stage(buf ^ 1, k0 + 32);  // prefetch next tile

    v16bf af[2], bfr[2];
#pragma unroll
    for (int mi = 0; mi < 2; ++mi) {
      const bf16* ap = &As[buf][wm * 32 + mi * 16 + l16][hl * 8];
      af[mi] = frag2(ap, ap + 16);   // A layout: K kb..kb+7, kb+16..kb+23
    }
#pragma unroll
    for (int ni = 0; ni < 2; ++ni) {
      const bf16* bp = &Bs[buf][wn * 32 + ni * 16 + l16][hl * 8];
      bfr[ni] = frag2(bp, bp + 16);  // B layout: K kb..kb+7, kb+16..kb+23
    }
#pragma unroll
    for (int mi = 0; mi < 2; ++mi)
#pragma unroll
      for (int ni = 0; ni < 2; ++ni)
        acc[mi][ni] = __builtin_amdgcn_wmma_f32_16x16x32_bf16(
            false, af[mi], false, bfr[ni], (short)0, acc[mi][ni], false, false);

    wait_stage();
    __syncthreads();
    buf ^= 1;
  }

  // epilogue: C layout row = r + 8*hl, col = l16
#pragma unroll
  for (int mi = 0; mi < 2; ++mi)
#pragma unroll
    for (int ni = 0; ni < 2; ++ni)
#pragma unroll
      for (int r = 0; r < 8; ++r) {
        int row = row0 + wm * 32 + mi * 16 + r + hl * 8;
        int col = n0 + wn * 32 + ni * 16 + l16;
        size_t idx = (size_t)row * N + col;
        float v = acc[mi][ni][r] + (bias ? bias[col] : 0.f);
        if (MODE == 0) {
          ((bf16*)outv)[idx] = (bf16)v;
        } else if (MODE == 1) {
          float gel = 0.5f * v * (1.f + erff(v * 0.70710678118654752f));
          ((bf16*)outv)[idx] = (bf16)gel;
        } else if (MODE == 2) {
          ((float*)outv)[idx] = resid[idx] + v;
        } else {
          ((float*)outv)[idx] = v;
        }
      }
}

// ---------------------------------------------------------------------------
// Flash attention: block = 128 threads (4 waves) handles (b, h, 64 q-rows).
// Each wave owns 16 q-rows. S = Q K^T (WMMA), online softmax, O += P V (WMMA).
// q/k are bf16 [B*T, D] (head h at column h*64); vt is bf16 [B,H,DH,T].
// ---------------------------------------------------------------------------
__global__ __launch_bounds__(128)
void attn_kernel(const bf16* __restrict__ q, const bf16* __restrict__ k,
                 const bf16* __restrict__ vt, bf16* __restrict__ y) {
  const int b = blockIdx.z, hh = blockIdx.y, qt0 = blockIdx.x * 64;
  const int tid = threadIdx.x, w = tid >> 5, lane = tid & 31;
  const int l16 = lane & 15, hl = lane >> 4;

  __shared__ bf16 Qs[64][72];      // [q-row][dh]
  __shared__ bf16 Ks[64][72];      // [key][dh] (row-major == B^T for S)
  __shared__ bf16 VsT[64][72];     // [dh][key] (from pre-transposed vt)
  __shared__ bf16 Ps[4][16][72];   // per-wave P staging (C->A relayout)

  const size_t base  = ((size_t)b * TT) * DM + (size_t)hh * DH;
  const size_t vbase = ((size_t)(b * HH + hh)) * DH * TT;

#pragma unroll
  for (int j = 0; j < 4; ++j) {    // Q tile 64x64
    int chunk = tid + j * 128;     // 0..511
    int row = chunk >> 3, c8 = (chunk & 7) * 8;
    stage_b128(&q[base + (size_t)(qt0 + row) * DM + c8], &Qs[row][c8]);
  }

  v8f z = {};
  v8f o[4]; o[0] = z; o[1] = z; o[2] = z; o[3] = z;
  float mrow[8], lrow[8];
#pragma unroll
  for (int r = 0; r < 8; ++r) { mrow[r] = -1e30f; lrow[r] = 0.f; }

  for (int kt = 0; kt < TT / 64; ++kt) {
#pragma unroll
    for (int j = 0; j < 4; ++j) {  // K tile (row-major) + V tile (pre-transposed)
      int chunk = tid + j * 128;
      int row = chunk >> 3, c8 = (chunk & 7) * 8;
      stage_b128(&k[base + (size_t)(kt * 64 + row) * DM + c8], &Ks[row][c8]);
      stage_b128(&vt[vbase + (size_t)row * TT + kt * 64 + c8], &VsT[row][c8]);
    }
    wait_stage();
    __syncthreads();

    // S = Q K^T * 1/sqrt(DH): K-dim = dh (64) -> 2 steps, 4 n-tiles (keys)
    v8f s[4]; s[0] = z; s[1] = z; s[2] = z; s[3] = z;
#pragma unroll
    for (int step = 0; step < 2; ++step) {
      const bf16* ap = &Qs[w * 16 + l16][step * 32 + hl * 8];
      v16bf af = frag2(ap, ap + 16);
#pragma unroll
      for (int ni = 0; ni < 4; ++ni) {
        const bf16* bp = &Ks[ni * 16 + l16][step * 32 + hl * 8];
        v16bf bf_ = frag2(bp, bp + 16);
        s[ni] = __builtin_amdgcn_wmma_f32_16x16x32_bf16(
            false, af, false, bf_, (short)0, s[ni], false, false);
      }
    }
#pragma unroll
    for (int ni = 0; ni < 4; ++ni) s[ni] = s[ni] * 0.125f;  // 1/sqrt(64)

    // online softmax per row (row r lives in one 16-lane half)
#pragma unroll
    for (int r = 0; r < 8; ++r) {
      float tm = fmaxf(fmaxf(s[0][r], s[1][r]), fmaxf(s[2][r], s[3][r]));
#pragma unroll
      for (int off = 1; off < 16; off <<= 1) tm = fmaxf(tm, __shfl_xor(tm, off, 16));
      float mnew = fmaxf(mrow[r], tm);
      float scl = __expf(mrow[r] - mnew);
      mrow[r] = mnew;
      float rs = 0.f;
#pragma unroll
      for (int ni = 0; ni < 4; ++ni) {
        float p = __expf(s[ni][r] - mnew);
        s[ni][r] = p; rs += p;
      }
#pragma unroll
      for (int off = 1; off < 16; off <<= 1) rs += __shfl_xor(rs, off, 16);
      lrow[r] = lrow[r] * scl + rs;
#pragma unroll
      for (int ni = 0; ni < 4; ++ni) o[ni][r] *= scl;
#pragma unroll
      for (int ni = 0; ni < 4; ++ni)
        Ps[w][r + hl * 8][ni * 16 + l16] = (bf16)s[ni][r];
    }
    __syncthreads();

    // O += P V : K-dim = keys (64) -> 2 steps; B from VsT, n-tiles over dh
#pragma unroll
    for (int step = 0; step < 2; ++step) {
      const bf16* ap = &Ps[w][l16][step * 32 + hl * 8];
      v16bf af = frag2(ap, ap + 16);
#pragma unroll
      for (int ni = 0; ni < 4; ++ni) {
        const bf16* bp = &VsT[ni * 16 + l16][step * 32 + hl * 8];
        v16bf bf_ = frag2(bp, bp + 16);
        o[ni] = __builtin_amdgcn_wmma_f32_16x16x32_bf16(
            false, af, false, bf_, (short)0, o[ni], false, false);
      }
    }
    __syncthreads();
  }

  // y = O / l
#pragma unroll
  for (int ni = 0; ni < 4; ++ni)
#pragma unroll
    for (int r = 0; r < 8; ++r) {
      int row = qt0 + w * 16 + r + hl * 8;
      int col = hh * DH + ni * 16 + l16;
      y[((size_t)(b * TT + row)) * DM + col] = (bf16)(o[ni][r] / lrow[r]);
    }
}

// ---------------------------------------------------------------------------
// Orchestration
// ---------------------------------------------------------------------------
static inline void cvtT(const float* src, bf16* dst, int K, int N, hipStream_t s) {
  cvtT_kernel<<<dim3(N / 32, K / 32), 256, 0, s>>>(src, dst, K, N);
}

extern "C" void kernel_launch(void* const* d_in, const int* in_sizes, int n_in,
                              void* d_out, int out_size, void* d_ws, size_t ws_size,
                              hipStream_t stream) {
  const float* seq    = (const float*)d_in[0];
  const float* pos    = (const float*)d_in[1];
  const float* ln1_g  = (const float*)d_in[2];
  const float* ln1_b  = (const float*)d_in[3];
  const float* wq     = (const float*)d_in[4];
  const float* bq     = (const float*)d_in[5];
  const float* wk     = (const float*)d_in[6];
  const float* bk     = (const float*)d_in[7];
  const float* wv     = (const float*)d_in[8];
  const float* bv     = (const float*)d_in[9];
  const float* wo     = (const float*)d_in[10];
  const float* bo     = (const float*)d_in[11];
  const float* ln2_g  = (const float*)d_in[12];
  const float* ln2_b  = (const float*)d_in[13];
  const float* w1     = (const float*)d_in[14];
  const float* b1     = (const float*)d_in[15];
  const float* w2     = (const float*)d_in[16];
  const float* b2     = (const float*)d_in[17];
  const float* lnf_g  = (const float*)d_in[18];
  const float* lnf_b  = (const float*)d_in[19];
  const float* w_head = (const float*)d_in[20];

  char* ws = (char*)d_ws;
  float* x  = (float*)(ws);                           // 16 MB f32 residual
  bf16*  h  = (bf16*)(ws + (size_t)16  * (1 << 20));  //  8 MB LN output
  bf16*  qb = (bf16*)(ws + (size_t)24  * (1 << 20));  //  8 MB
  bf16*  kb = (bf16*)(ws + (size_t)32  * (1 << 20));  //  8 MB
  bf16*  vb = (bf16*)(ws + (size_t)40  * (1 << 20));  //  8 MB
  bf16*  yb = (bf16*)(ws + (size_t)48  * (1 << 20));  //  8 MB
  bf16*  a1 = (bf16*)(ws + (size_t)56  * (1 << 20));  // 32 MB FFN hidden
  bf16*  wb = (bf16*)(ws + (size_t)88  * (1 << 20));  //  8 MB bf16 W^T scratch
  bf16*  vt = (bf16*)(ws + (size_t)96  * (1 << 20));  //  8 MB transposed V

  long totXD = (long)MR * DM;
  addpos_kernel<<<(int)((totXD + 255) / 256), 256, 0, stream>>>(seq, pos, x, totXD);

  dim3 gD(DM / 64, MR / 128);   // N=1024
  dim3 gF(FF / 64, MR / 128);   // N=4096
  dim3 gA(TT / 64, HH, BB);
  dim3 gV(TT / 32, DH / 32, BB * HH);

  for (int l = 0; l < LL; ++l) {
    ln_kernel<<<MR, 256, 0, stream>>>(x, ln1_g + l * DM, ln1_b + l * DM, h);

    cvtT(wq + (size_t)l * DM * DM, wb, DM, DM, stream);
    gemm_kernel<0><<<gD, 256, 0, stream>>>(h, wb, bq + l * DM, qb, nullptr, MR, DM, DM);
    cvtT(wk + (size_t)l * DM * DM, wb, DM, DM, stream);
    gemm_kernel<0><<<gD, 256, 0, stream>>>(h, wb, bk + l * DM, kb, nullptr, MR, DM, DM);
    cvtT(wv + (size_t)l * DM * DM, wb, DM, DM, stream);
    gemm_kernel<0><<<gD, 256, 0, stream>>>(h, wb, bv + l * DM, vb, nullptr, MR, DM, DM);

    vtrans_kernel<<<gV, 256, 0, stream>>>(vb, vt);
    attn_kernel<<<gA, 128, 0, stream>>>(qb, kb, vt, yb);

    cvtT(wo + (size_t)l * DM * DM, wb, DM, DM, stream);
    gemm_kernel<2><<<gD, 256, 0, stream>>>(yb, wb, bo + l * DM, x, x, MR, DM, DM);

    ln_kernel<<<MR, 256, 0, stream>>>(x, ln2_g + l * DM, ln2_b + l * DM, h);

    cvtT(w1 + (size_t)l * DM * FF, wb, DM, FF, stream);
    gemm_kernel<1><<<gF, 256, 0, stream>>>(h, wb, b1 + l * FF, a1, nullptr, MR, FF, DM);
    cvtT(w2 + (size_t)l * FF * DM, wb, FF, DM, stream);
    gemm_kernel<2><<<gD, 256, 0, stream>>>(a1, wb, b2 + l * DM, x, x, MR, DM, FF);
  }

  ln_kernel<<<MR, 256, 0, stream>>>(x, lnf_g, lnf_b, h);
  cvtT(w_head, wb, DM, OUTD, stream);
  gemm_kernel<3><<<dim3(OUTD / 64, MR / 128), 256, 0, stream>>>(
      h, wb, nullptr, d_out, nullptr, MR, OUTD, DM);
}
